// AffinityHead_3556232921658
// MI455X (gfx1250) — compile-verified, matrix operand order checked by reading, not run on hardware
//
#include <hip/hip_runtime.h>
#include <hip/hip_bf16.h>

#define N_IMG 2
#define NBOX  64
#define NV    2560
#define NE    5120
#define NL    64
#define HD    128
#define HH    256

typedef __attribute__((ext_vector_type(16))) __bf16 v16bf;
typedef __attribute__((ext_vector_type(8)))  __bf16 v8bf;
typedef __attribute__((ext_vector_type(8)))  float  v8f;

__device__ __forceinline__ float siluf(float x) { return x / (1.f + __expf(-x)); }
__device__ __forceinline__ float sigmf(float x) { return 1.f / (1.f + __expf(-x)); }
__device__ __forceinline__ float lrelu(float x) { return x > 0.f ? x : 0.01f * x; }

// WMMA 16-bit fragment (ISA 7.12.2): lane group g needs k = g*8+[0..7] and
// g*8+[16..23] of each 32-wide k-block -> two contiguous 16B chunks in a
// row-major bf16 row => two ds_load_b128 per fragment.
__device__ __forceinline__ v16bf load_frag(const __bf16* rowp, int kk, int g) {
  const v8bf lo = *(const v8bf*)(rowp + kk * 32 + g * 8);
  const v8bf hi = *(const v8bf*)(rowp + kk * 32 + g * 8 + 16);
  return __builtin_shufflevector(lo, hi, 0, 1, 2, 3, 4, 5, 6, 7,
                                 8, 9, 10, 11, 12, 13, 14, 15);
}

// ---------------------------------------------------------------------------
// 1) pocket path: S[n,s,c,k] = sum over valid conv positions of silu(x)[c, p+k]
//    (mean-of-conv == conv-weights dot windowed-sums / P)
// ---------------------------------------------------------------------------
__global__ void k_silu_wsum(const float* m0, const float* m1, const float* m2,
                            const float* m3, const float* m4, float* S) {
  const int sps[5] = {48, 24, 12, 6, 3};
  int bi = blockIdx.x;
  int c  = bi % 96;
  int s  = (bi / 96) % 5;
  int n  = bi / (96 * 5);
  const float* ms = (s == 0) ? m0 : (s == 1) ? m1 : (s == 2) ? m2 : (s == 3) ? m3 : m4;
  int sp = sps[s], out = sp - 2;
  int sp2 = sp * sp, sp3 = sp2 * sp;
  const float* base = ms + ((size_t)n * 96 + c) * sp3;
  float acc[27];
#pragma unroll
  for (int k = 0; k < 27; k++) acc[k] = 0.f;
  for (int idx = threadIdx.x; idx < sp3; idx += blockDim.x) {
    int id = idx / sp2, ih = (idx / sp) % sp, iw = idx % sp;
    float v = siluf(base[idx]);
#pragma unroll
    for (int kd = 0; kd < 3; kd++) {
      int od = id - kd; if (od < 0 || od >= out) continue;
#pragma unroll
      for (int kh = 0; kh < 3; kh++) {
        int oh = ih - kh; if (oh < 0 || oh >= out) continue;
#pragma unroll
        for (int kw = 0; kw < 3; kw++) {
          int ow = iw - kw; if (ow < 0 || ow >= out) continue;
          acc[(kd * 3 + kh) * 3 + kw] += v;
        }
      }
    }
  }
  __shared__ float red[27 * 256];
  for (int k = 0; k < 27; k++) red[k * 256 + threadIdx.x] = acc[k];
  __syncthreads();
  for (int st = 128; st > 0; st >>= 1) {
    if (threadIdx.x < st)
      for (int k = 0; k < 27; k++)
        red[k * 256 + threadIdx.x] += red[k * 256 + threadIdx.x + st];
    __syncthreads();
  }
  if (threadIdx.x < 27)
    S[(((size_t)n * 5 + s) * 96 + c) * 27 + threadIdx.x] = red[threadIdx.x * 256];
}

__global__ void k_pocket_feats(const float* conv_w, const float* conv_b,
                               const float* S, float* feats) {
  const int outs[5] = {46, 22, 10, 4, 1}; // per scale s
  int bi = blockIdx.x;       // n*5 + i
  int n = bi / 5, i = bi % 5;
  int o = threadIdx.x;       // 256
  int s = 4 - i;             // conv_w[i] applies to ms[4-i]
  float P = (float)outs[s]; P = P * P * P;
  const float* w  = conv_w + ((size_t)i * HH + o) * 96 * 27;
  const float* Sn = S + (((size_t)n * 5 + s) * 96) * 27;
  float acc = 0.f;
  for (int ck = 0; ck < 96 * 27; ck++) acc += w[ck] * Sn[ck];
  feats[(size_t)n * 1280 + i * HH + o] = acc / P + conv_b[i * HH + o];
}

__global__ void k_pocket_head(const float* feats, const float* W, const float* bias,
                              float* pocket) {
  __shared__ float sf[1280];
  int n = blockIdx.x, o = threadIdx.x;
  for (int k = threadIdx.x; k < 1280; k += blockDim.x)
    sf[k] = siluf(feats[(size_t)n * 1280 + k]);
  __syncthreads();
  float acc = bias[o];
  for (int k = 0; k < 1280; k++) acc += sf[k] * W[(size_t)k * HH + o];
  pocket[n * HH + o] = acc;
}

// ---------------------------------------------------------------------------
// 2) token path
// ---------------------------------------------------------------------------
__global__ void k_token(const float* tokf, const float* W, const float* bias, float* tf) {
  __shared__ float st[192];
  int bi = blockIdx.x;  // n*NBOX + b
  int t  = threadIdx.x; // 256
  if (t < 192) st[t] = siluf(tokf[(size_t)bi * 192 + t]);
  __syncthreads();
  float s = bias[t];
  for (int k = 0; k < 192; k++) s += st[k] * W[(size_t)k * HH + t];
  tf[(size_t)bi * HH + t] = s;
}

__global__ void k_pfgate(const float* tf, const float* pocket,
                         const float* cat_w, const float* cat_b,
                         const float* gate_w, const float* gate_b, float* pf) {
  __shared__ float pb[768];
  int n = blockIdx.x, h = threadIdx.x; // 256
  float s = 0.f;
  for (int b = 0; b < NBOX; b++) s += tf[((size_t)n * NBOX + b) * HH + h];
  pb[h]          = pocket[n * HH + h];
  pb[HH + h]     = s;
  pb[2 * HH + h] = s / (float)NBOX;
  __syncthreads();
  float c = cat_b[h], g = gate_b[h];
  for (int k = 0; k < 768; k++) {
    float p = pb[k];
    c += p * cat_w[(size_t)k * HH + h];
    g += p * gate_w[(size_t)k * HH + h];
  }
  pf[n * HH + h] = c * sigmf(g);
}

// ---------------------------------------------------------------------------
// 3) graph path
// ---------------------------------------------------------------------------
__global__ void k_embed_nodes(const int* node_x, const float* emb, float* x0, float* xa) {
  size_t i = (size_t)blockIdx.x * blockDim.x + threadIdx.x;
  if (i >= (size_t)N_IMG * NV * HD) return;
  int f = i % HD;
  int v = (int)((i / HD) % NV);
  int n = (int)(i / ((size_t)NV * HD));
  float val = emb[(size_t)node_x[n * NV + v] * HD + f];
  x0[i] = val;
  xa[i] = val;
}

__global__ void k_embed_edges(const int* ea, const float* emb, float* ef) {
  size_t i = (size_t)blockIdx.x * blockDim.x + threadIdx.x;
  if (i >= (size_t)N_IMG * NE * HD) return;
  int f = i % HD;
  int e = (int)((i / HD) % NE);
  int n = (int)(i / ((size_t)NE * HD));
  ef[i] = emb[(size_t)ea[n * NE + e] * HD + f];
}

// deterministic CSR (stable counting sort by dst) + graph node ranges
__global__ void k_csr(const int* edge_index, const int* bv,
                      int* off, int* fill, int* csr, int* gs) {
  int n = blockIdx.x;
  const int* dst = edge_index + (size_t)n * 2 * NE + NE;
  int* offn  = off  + n * (NV + 1);
  int* filln = fill + n * NV;
  int* csrn  = csr  + n * NE;
  for (int v = 0; v <= NV; v++) offn[v] = 0;
  for (int e = 0; e < NE; e++) offn[dst[e] + 1]++;
  for (int v = 0; v < NV; v++) offn[v + 1] += offn[v];
  for (int v = 0; v < NV; v++) filln[v] = 0;
  for (int e = 0; e < NE; e++) { int d = dst[e]; csrn[offn[d] + filln[d]++] = e; }
  const int* bvn = bv + n * NV;
  int* gsn = gs + n * (NL + 1);
  int ptr = 0;
  for (int l = 0; l <= NL; l++) {
    while (ptr < NV && bvn[ptr] < l) ptr++;
    gsn[l] = ptr;
  }
}

// y[v] = x[v] + sum_{e: dst=v} relu(x[src[e]] + efeat[e])   (GINE, eps=0)
__global__ void k_gine_agg(const float* x, const float* ef, const int* edge_index,
                           const int* off, const int* csr, float* y) {
  size_t i = (size_t)blockIdx.x * blockDim.x + threadIdx.x;
  if (i >= (size_t)N_IMG * NV * HD) return;
  int f = i % HD;
  int v = (int)((i / HD) % NV);
  int n = (int)(i / ((size_t)NV * HD));
  const int* src  = edge_index + (size_t)n * 2 * NE;
  const int* offn = off + n * (NV + 1);
  const int* csrn = csr + n * NE;
  const float* xn  = x  + (size_t)n * NV * HD;
  const float* efn = ef + (size_t)n * NE * HD;
  float a = xn[(size_t)v * HD + f];
  for (int p = offn[v]; p < offn[v + 1]; p++) {
    int e = csrn[p];
    float m = xn[(size_t)src[e] * HD + f] + efn[(size_t)e * HD + f];
    a += (m > 0.f) ? m : 0.f;
  }
  y[i] = a;
}

// ---------------------------------------------------------------------------
// generic bf16 WMMA GEMM: C[M,N] = act(A[M,K] @ W[K,N] + bias), row-major.
// K,N compile-time -> static acc indexing + fully unrolled wmma chains.
// block: 128 threads = 4 waves; one 16-row M tile / block; N split by wave.
// W staged TRANSPOSED in LDS so B fragments are two ds_load_b128 each.
// ---------------------------------------------------------------------------
template <int K, int N, int ACT>
__global__ __launch_bounds__(128) void k_wmma_gemm(const float* __restrict__ A,
                                                   const float* __restrict__ W,
                                                   const float* __restrict__ bias,
                                                   float* __restrict__ C) {
  constexpr int NT = (N / 16) / 4; // tiles per wave
  extern __shared__ char smem[];
  __bf16* sBt = (__bf16*)smem;                       // [N][K]  (W transposed)
  __bf16* sA  = (__bf16*)(smem + (size_t)N * K * 2); // [16][K]
  int tid = threadIdx.x;
  int m0  = blockIdx.x * 16;
  for (int i = tid; i < K * N; i += 128) {
    int k = i / N, n = i % N;
    sBt[n * K + k] = (__bf16)W[i];
  }
  for (int i = tid; i < 16 * K; i += 128) {
    int r = i / K, k = i % K;
    sA[i] = (__bf16)A[(size_t)(m0 + r) * K + k];
  }
  __syncthreads();
  int lane = tid & 31, wave = tid >> 5;
  int col = lane & 15, g = lane >> 4;
  v8f zero = {0.f, 0.f, 0.f, 0.f, 0.f, 0.f, 0.f, 0.f};
  v8f acc[NT];
#pragma unroll
  for (int t = 0; t < NT; t++) acc[t] = zero;
#pragma unroll
  for (int kk = 0; kk < K / 32; kk++) {
    v16bf a = load_frag(sA + col * K, kk, g);
#pragma unroll
    for (int t = 0; t < NT; t++) {
      int n0 = (wave * NT + t) * 16;
      v16bf b = load_frag(sBt + (size_t)(n0 + col) * K, kk, g);
      acc[t] = __builtin_amdgcn_wmma_f32_16x16x32_bf16(
          false, a, false, b, (short)0, acc[t], false, false);
    }
  }
#pragma unroll
  for (int t = 0; t < NT; t++) {
    int n = (wave * NT + t) * 16 + col;
    float bn = bias[n];
#pragma unroll
    for (int r = 0; r < 8; r++) {
      int m = m0 + r + g * 8;
      float v = acc[t][r] + bn;
      if (ACT == 1) v = lrelu(v);
      C[(size_t)m * N + n] = v;
    }
  }
}

__global__ void k_add(const float* a, const float* b, float* o, int nelem) {
  int i = blockIdx.x * blockDim.x + threadIdx.x;
  if (i < nelem) o[i] = a[i] + b[i];
}

__global__ void k_mul(const float* a, const float* b, float* o, int nelem) {
  int i = blockIdx.x * blockDim.x + threadIdx.x;
  if (i < nelem) o[i] = a[i] * b[i];
}

__global__ void k_layernorm(const float* Hin, const float* gg, const float* bb,
                            float* Xout) {
  __shared__ float red[HH];
  int row = blockIdx.x, t = threadIdx.x; // 256 threads
  float v = Hin[(size_t)row * HH + t];
  red[t] = v;
  __syncthreads();
  for (int st = 128; st > 0; st >>= 1) { if (t < st) red[t] += red[t + st]; __syncthreads(); }
  float mean = red[0] / (float)HH;
  __syncthreads();
  float d = v - mean;
  red[t] = d * d;
  __syncthreads();
  for (int st = 128; st > 0; st >>= 1) { if (t < st) red[t] += red[t + st]; __syncthreads(); }
  float var = red[0] / (float)HH;
  Xout[(size_t)row * HH + t] = d * rsqrtf(var + 1e-5f) * gg[t] + bb[t];
}

// Z[n,l,:] = [sum_v xs, sum/clip(cnt,1)] using sorted batch ranges
__global__ void k_pool(const float* xs, const int* gs, float* Z) {
  int bi = blockIdx.x;          // n*NL + l
  int n = bi / NL, l = bi % NL;
  int f = threadIdx.x;          // 128
  const int* gsn = gs + n * (NL + 1);
  int a = gsn[l], e = gsn[l + 1];
  float s = 0.f;
  for (int v = a; v < e; v++) s += xs[((size_t)n * NV + v) * HD + f];
  float cnt = (float)(e - a); if (cnt < 1.f) cnt = 1.f;
  Z[((size_t)n * NL + l) * HH + f]      = s;
  Z[((size_t)n * NL + l) * HH + HD + f] = s / cnt;
}

// ---------------------------------------------------------------------------
// 4) interaction: per (n, 16-row v-tile): loop b over boxes,
//    A = X_tile ⊙ tf[b] (staged to LDS as bf16), C = leaky(A @ imlp_w + b),
//    pe_v[v] += (C·pe_w+pe_b)*sigmoid(C·pg_w+pg_b).  imlp_w^T staged once.
// ---------------------------------------------------------------------------
__global__ __launch_bounds__(128) void k_interact(
    const float* __restrict__ X, const float* __restrict__ tf,
    const float* __restrict__ W, const float* __restrict__ bias,
    const float* __restrict__ pe_w, const float* __restrict__ pe_b,
    const float* __restrict__ pg_w, const float* __restrict__ pg_b,
    float* __restrict__ pe_v) {
  extern __shared__ char smem[];
  __bf16* sWt = (__bf16*)smem;                            // [256][256] W^T bf16
  __bf16* sAb = (__bf16*)(smem + HH * HH * 2);            // [16][256] bf16
  float* sX   = (float*)(smem + HH * HH * 2 + 16 * HH * 2); // 16*256 f32
  float* sC   = sX + 16 * HH;                             // 16*256
  float* stf  = sC + 16 * HH;                             // 256
  float* sbias = stf + HH;                                // 256
  float* spe  = sbias + HH;                               // 256
  float* spg  = spe + HH;                                 // 256
  float* pd1  = spg + HH;                                 // 128
  float* pd2  = pd1 + 128;                                // 128
  float* srow = pd2 + 128;                                // 16
  int tid = threadIdx.x; // 128
  int n  = blockIdx.y;
  int v0 = blockIdx.x * 16;
  for (int i = tid; i < HH * HH; i += 128) {
    int k = i / HH, nn = i % HH;
    sWt[nn * HH + k] = (__bf16)W[i];
  }
  for (int i = tid; i < 16 * HH; i += 128)
    sX[i] = X[((size_t)n * NV + v0) * HH + i];
  for (int i = tid; i < HH; i += 128) {
    sbias[i] = bias[i];
    spe[i] = pe_w[i];
    spg[i] = pg_w[i];
  }
  if (tid < 16) srow[tid] = 0.f;
  __syncthreads();
  int lane = tid & 31, wave = tid >> 5;
  int col = lane & 15, g = lane >> 4;
  int row8 = tid >> 3, seg = tid & 7;
  float peb = pe_b[0], pgb = pg_b[0];
  v8f zero = {0.f, 0.f, 0.f, 0.f, 0.f, 0.f, 0.f, 0.f};
  for (int b = 0; b < NBOX; b++) {
    for (int i = tid; i < HH; i += 128)
      stf[i] = tf[((size_t)n * NBOX + b) * HH + i];
    if (tid == 0 && b + 1 < NBOX)
      __builtin_prefetch(&tf[((size_t)n * NBOX + b + 1) * HH], 0, 0); // global_prefetch_b8
    __syncthreads();
    // scaled A tile -> bf16 LDS (pairs, 32-bit stores)
    for (int idx = tid; idx < 16 * HH / 2; idx += 128) {
      int i2 = idx * 2;
      int k = i2 & (HH - 1);
      union { __bf16 h[2]; unsigned u; } pk;
      pk.h[0] = (__bf16)(sX[i2] * stf[k]);
      pk.h[1] = (__bf16)(sX[i2 + 1] * stf[k + 1]);
      ((unsigned*)sAb)[idx] = pk.u;
    }
    __syncthreads();
    v8f acc[4];
#pragma unroll
    for (int t = 0; t < 4; t++) acc[t] = zero;
#pragma unroll
    for (int kk = 0; kk < HH / 32; kk++) {
      v16bf a = load_frag(sAb + col * HH, kk, g);
#pragma unroll
      for (int t = 0; t < 4; t++) {
        int n0 = (wave * 4 + t) * 16;
        v16bf bb = load_frag(sWt + (size_t)(n0 + col) * HH, kk, g);
        acc[t] = __builtin_amdgcn_wmma_f32_16x16x32_bf16(
            false, a, false, bb, (short)0, acc[t], false, false);
      }
    }
#pragma unroll
    for (int t = 0; t < 4; t++) {
      int nn = (wave * 4 + t) * 16 + col;
#pragma unroll
      for (int r = 0; r < 8; r++) sC[(r + g * 8) * HH + nn] = acc[t][r];
    }
    __syncthreads();
    // epilogue: deterministic 8-way split reduction per row
    {
      float d1 = 0.f, d2 = 0.f;
      int h0 = seg * 32;
      for (int h = h0; h < h0 + 32; h++) {
        float val = lrelu(sC[row8 * HH + h] + sbias[h]);
        d1 += val * spe[h];
        d2 += val * spg[h];
      }
      pd1[tid] = d1;
      pd2[tid] = d2;
      __syncthreads();
      for (int st = 4; st > 0; st >>= 1) {
        if (seg < st) { pd1[tid] += pd1[tid + st]; pd2[tid] += pd2[tid + st]; }
        __syncthreads();
      }
      if (seg == 0)
        srow[row8] += (pd1[tid] + peb) * sigmf(pd2[tid] + pgb);
      __syncthreads();
    }
  }
  if (tid < 16) pe_v[(size_t)n * NV + v0 + tid] = srow[tid];
}

// ---------------------------------------------------------------------------
// 5) final: out[n,l] = sum_{v in graph l} pe_v + eb2( leaky(eb1([pf, Zg])) )
// ---------------------------------------------------------------------------
__global__ void k_out(const float* pf, const float* Zg, const float* pe_v,
                      const int* gs, const float* eb1_w, const float* eb1_b,
                      const float* eb2_w, const float* eb2_b, float* out) {
  __shared__ float pb[512];
  __shared__ float red[HH];
  __shared__ float red2[HH];
  int bi = blockIdx.x;          // n*NL + l
  int n = bi / NL, l = bi % NL;
  int t = threadIdx.x;          // 256
  pb[t]      = pf[n * HH + t];
  pb[HH + t] = Zg[(size_t)bi * HH + t];
  __syncthreads();
  float hid = eb1_b[t];
  for (int k = 0; k < 512; k++) hid += pb[k] * eb1_w[(size_t)k * HH + t];
  hid = lrelu(hid);
  red[t] = hid * eb2_w[t];
  const int* gsn = gs + n * (NL + 1);
  float pp = 0.f;
  for (int v = gsn[l] + t; v < gsn[l + 1]; v += blockDim.x)
    pp += pe_v[(size_t)n * NV + v];
  red2[t] = pp;
  __syncthreads();
  for (int st = 128; st > 0; st >>= 1) {
    if (t < st) { red[t] += red[t + st]; red2[t] += red2[t + st]; }
    __syncthreads();
  }
  if (t == 0) out[bi] = red2[0] + red[0] + eb2_b[0];
}

// ---------------------------------------------------------------------------
extern "C" void kernel_launch(void* const* d_in, const int* in_sizes, int n_in,
                              void* d_out, int out_size, void* d_ws, size_t ws_size,
                              hipStream_t stream) {
  (void)in_sizes; (void)n_in; (void)out_size; (void)ws_size;
  const float* ms0  = (const float*)d_in[0];
  const float* ms1  = (const float*)d_in[1];
  const float* ms2  = (const float*)d_in[2];
  const float* ms3  = (const float*)d_in[3];
  const float* ms4  = (const float*)d_in[4];
  const float* tokf = (const float*)d_in[5];
  const int* node_x     = (const int*)d_in[6];
  const int* edge_attr  = (const int*)d_in[7];
  const int* edge_index = (const int*)d_in[8];
  const int* batch_vec  = (const int*)d_in[9];
  const float* node_emb = (const float*)d_in[11];
  const float* edge_emb = (const float*)d_in[12];
  const float* gine_w = (const float*)d_in[13];
  const float* gine_b = (const float*)d_in[14];
  const float* head_w = (const float*)d_in[15];
  const float* head_b = (const float*)d_in[16];
  const float* ln_g = (const float*)d_in[17];
  const float* ln_b = (const float*)d_in[18];
  const float* ro_w = (const float*)d_in[19];
  const float* ro_b = (const float*)d_in[20];
  const float* rg_w = (const float*)d_in[21];
  const float* rg_b = (const float*)d_in[22];
  const float* token_w = (const float*)d_in[23];
  const float* token_b = (const float*)d_in[24];
  const float* conv_w = (const float*)d_in[25];
  const float* conv_b = (const float*)d_in[26];
  const float* pocket_w = (const float*)d_in[27];
  const float* pocket_b = (const float*)d_in[28];
  const float* cat_w  = (const float*)d_in[29];
  const float* cat_b  = (const float*)d_in[30];
  const float* gate_w = (const float*)d_in[31];
  const float* gate_b = (const float*)d_in[32];
  const float* imlp_w = (const float*)d_in[33];
  const float* imlp_b = (const float*)d_in[34];
  const float* pe_w = (const float*)d_in[35];
  const float* pe_b = (const float*)d_in[36];
  const float* pg_w = (const float*)d_in[37];
  const float* pg_b = (const float*)d_in[38];
  const float* eb1_w = (const float*)d_in[39];
  const float* eb1_b = (const float*)d_in[40];
  const float* eb2_w = (const float*)d_in[41];
  const float* eb2_b = (const float*)d_in[42];
  float* out = (float*)d_out;

  char* ws = (char*)d_ws;
  size_t off = 0;
  auto carve = [&](size_t bytes) -> void* {
    void* p = ws + off;
    off += (bytes + 255) & ~(size_t)255;
    return p;
  };
  float* S      = (float*)carve((size_t)2 * 5 * 96 * 27 * 4);
  float* feats  = (float*)carve((size_t)2 * 1280 * 4);
  float* pocket = (float*)carve((size_t)2 * HH * 4);
  float* tf     = (float*)carve((size_t)2 * NBOX * HH * 4);
  float* pf     = (float*)carve((size_t)2 * HH * 4);
  float* x0     = (float*)carve((size_t)2 * NV * HD * 4);
  float* xa     = (float*)carve((size_t)2 * NV * HD * 4);
  float* ybuf   = (float*)carve((size_t)2 * NV * HD * 4);
  float* xs     = (float*)carve((size_t)2 * NV * HD * 4);
  float* efeat  = (float*)carve((size_t)2 * NE * HD * 4);
  int*   csrOff = (int*)carve((size_t)2 * (NV + 1) * 4);
  int*   csrFil = (int*)carve((size_t)2 * NV * 4);
  int*   csrArr = (int*)carve((size_t)2 * NE * 4);
  int*   gs     = (int*)carve((size_t)2 * (NL + 1) * 4);
  float* Hbuf   = (float*)carve((size_t)2 * NV * HH * 4);
  float* Xn     = (float*)carve((size_t)2 * NV * HH * 4);
  float* Z      = (float*)carve((size_t)2 * NL * HH * 4);
  float* Zr     = (float*)carve((size_t)2 * NL * HH * 4);
  float* Zo     = (float*)carve((size_t)2 * NL * HH * 4);
  float* Zg     = (float*)carve((size_t)2 * NL * HH * 4);
  float* pe_v   = (float*)carve((size_t)2 * NV * 4);

  // pocket + token paths
  k_silu_wsum<<<2 * 5 * 96, 256, 0, stream>>>(ms0, ms1, ms2, ms3, ms4, S);
  k_pocket_feats<<<2 * 5, 256, 0, stream>>>(conv_w, conv_b, S, feats);
  k_pocket_head<<<2, 256, 0, stream>>>(feats, pocket_w, pocket_b, pocket);
  k_token<<<2 * NBOX, 256, 0, stream>>>(tokf, token_w, token_b, tf);
  k_pfgate<<<2, 256, 0, stream>>>(tf, pocket, cat_w, cat_b, gate_w, gate_b, pf);

  // graph encoder
  k_embed_nodes<<<(2 * NV * HD) / 256, 256, 0, stream>>>(node_x, node_emb, x0, xa);
  k_embed_edges<<<(2 * NE * HD) / 256, 256, 0, stream>>>(edge_attr, edge_emb, efeat);
  k_csr<<<2, 1, 0, stream>>>(edge_index, batch_vec, csrOff, csrFil, csrArr, gs);

  size_t sh_gine = ((size_t)HD * HD + 16 * HD) * 2;
  for (int l = 0; l < 4; l++) {
    k_gine_agg<<<(2 * NV * HD) / 256, 256, 0, stream>>>(xa, efeat, edge_index,
                                                        csrOff, csrArr, ybuf);
    k_wmma_gemm<HD, HD, 1><<<(2 * NV) / 16, 128, sh_gine, stream>>>(
        ybuf, gine_w + (size_t)l * HD * HD, gine_b + (size_t)l * HD, xa);
  }
  k_add<<<(2 * NV * HD) / 256, 256, 0, stream>>>(x0, xa, xs, 2 * NV * HD);

  size_t sh_head = ((size_t)HD * HH + 16 * HD) * 2;
  k_wmma_gemm<HD, HH, 0><<<(2 * NV) / 16, 128, sh_head, stream>>>(xs, head_w,
                                                                  head_b, Hbuf);
  k_layernorm<<<2 * NV, 256, 0, stream>>>(Hbuf, ln_g, ln_b, Xn);

  k_pool<<<2 * NL, 128, 0, stream>>>(xs, gs, Z);
  size_t sh_z = ((size_t)HH * HH + 16 * HH) * 2;
  k_wmma_gemm<HH, HH, 0><<<(2 * NL) / 16, 128, sh_z, stream>>>(Z, rg_w, rg_b, Zr);
  k_wmma_gemm<HH, HH, 0><<<(2 * NL) / 16, 128, sh_z, stream>>>(Z, ro_w, ro_b, Zo);
  k_mul<<<(2 * NL * HH) / 256, 256, 0, stream>>>(Zr, Zo, Zg, 2 * NL * HH);

  // interaction (the big bf16 WMMA GEMM)
  size_t sh_int = (size_t)HH * HH * 2 + (size_t)16 * HH * 2 +
                  ((size_t)16 * HH + 16 * HH + HH * 4 + 128 + 128 + 16) * 4;
  dim3 ig(NV / 16, N_IMG);
  k_interact<<<ig, 128, sh_int, stream>>>(Xn, tf, imlp_w, imlp_b, pe_w, pe_b,
                                          pg_w, pg_b, pe_v);

  k_out<<<2 * NL, 256, 0, stream>>>(pf, Zg, pe_v, gs, eb1_w, eb1_b, eb2_w, eb2_b, out);
}
